// Joiner_20873541058896
// MI455X (gfx1250) — matrix operand order repaired and checked by reading, compiled, not run on hardware
//
#include <hip/hip_runtime.h>

typedef __attribute__((ext_vector_type(16))) _Float16 v16h;
typedef __attribute__((ext_vector_type(8)))  _Float16 v8h;
typedef __attribute__((ext_vector_type(4)))  _Float16 v4h;
typedef __attribute__((ext_vector_type(8)))  float    v8f;

#define B_ 8
#define T_ 256
#define U_ 64
#define D_ 512
#define V_ 1024

// Workgroup tile: M=128 (2 t-values x 64 u-values), N=128, K staged 64 at a time.
#define MT   128
#define NT   128
#define KT   64
#define ASTR 72   // f16 row stride in LDS (144B: 16B aligned, conflict-padded)
#define BSTR 72

__global__ __launch_bounds__(256)
void joiner_wmma_f16(const float* __restrict__ S, const float* __restrict__ G,
                     const float* __restrict__ W, const float* __restrict__ bias,
                     float* __restrict__ out) {
  __shared__ alignas(16) _Float16 ldsA[MT * ASTR];
  __shared__ alignas(16) _Float16 ldsB[NT * BSTR];

  const int tid     = threadIdx.x;      // 0..255, 8 waves of 32
  const int lane    = tid & 31;
  const int laneLow = lane & 15;        // row/col within 16x16 tile
  const int laneHi  = lane >> 4;        // which K half-slice this lane holds
  const int wave    = tid >> 5;
  const int wm      = wave & 1;         // M strip: rows [wm*64, wm*64+64)
  const int wn      = wave >> 1;        // N strip: cols [wn*32, wn*32+32)

  const int vBase = blockIdx.x * NT;
  const int t0    = blockIdx.y * 2;
  const int b     = blockIdx.z;

  // staging assignment: 2 threads per row, 32 floats each
  const int srow = tid >> 1;            // 0..127
  const int kOff = (tid & 1) * 32;      // 0 or 32
  const int tS   = t0 + (srow >> 6);
  const int uS   = srow & 63;

  const float* wptr = W + (size_t)(vBase + srow) * D_;
  const float* sptr = S + ((size_t)b * T_ + tS) * D_;
  const float* gptr = G + ((size_t)b * U_ + uS) * D_;

  v8f acc[4][2] = {};

  for (int k0 = 0; k0 < D_; k0 += KT) {
    // ---- stage B tile: W[vBase+row, k0..k0+63] -> f16 LDS ----
    {
      const float* wr = wptr + k0 + kOff;
      _Float16* bd = &ldsB[srow * BSTR + kOff];
#pragma unroll
      for (int j = 0; j < 8; ++j) {
        float4 w4 = reinterpret_cast<const float4*>(wr)[j];
        v4h h = { (_Float16)w4.x, (_Float16)w4.y, (_Float16)w4.z, (_Float16)w4.w };
        *reinterpret_cast<v4h*>(bd + j * 4) = h;
      }
      if (k0 + KT < D_) __builtin_prefetch(wr + KT, 0, 1);  // global_prefetch_b8
    }
    // ---- stage A tile: relu(s[t,:] + g[u,:]) -> f16 LDS ----
    {
      const float* sr = sptr + k0 + kOff;
      const float* gr = gptr + k0 + kOff;
      _Float16* ad = &ldsA[srow * ASTR + kOff];
#pragma unroll
      for (int j = 0; j < 8; ++j) {
        float4 s4 = reinterpret_cast<const float4*>(sr)[j];
        float4 g4 = reinterpret_cast<const float4*>(gr)[j];
        v4h h = { (_Float16)fmaxf(s4.x + g4.x, 0.0f),
                  (_Float16)fmaxf(s4.y + g4.y, 0.0f),
                  (_Float16)fmaxf(s4.z + g4.z, 0.0f),
                  (_Float16)fmaxf(s4.w + g4.w, 0.0f) };
        *reinterpret_cast<v4h*>(ad + j * 4) = h;
      }
    }
    __syncthreads();

#pragma unroll
    for (int st = 0; st < 2; ++st) {
      // B fragments: lane holds col n = base+laneLow, elems 0..7 -> K = st*32 + laneHi*8 + (0..7), elems 8..15 -> K = st*32 + 16 + laneHi*8 + (0..7)
      v16h bf[2];
#pragma unroll
      for (int nt = 0; nt < 2; ++nt) {
        const _Float16* bp =
            &ldsB[(wn * 32 + nt * 16 + laneLow) * BSTR + st * 32 + laneHi * 8];
        v8h lo = *reinterpret_cast<const v8h*>(bp);
        v8h hi = *reinterpret_cast<const v8h*>(bp + 16);
        bf[nt] = __builtin_shufflevector(lo, hi, 0, 1, 2, 3, 4, 5, 6, 7,
                                         8, 9, 10, 11, 12, 13, 14, 15);
      }
#pragma unroll
      for (int mt = 0; mt < 4; ++mt) {
        // A fragment: lane holds row m = base+laneLow,
        // elems 0..7  -> K = st*32 + laneHi*8 + (0..7)
        // elems 8..15 -> K = st*32 + 16 + laneHi*8 + (0..7)
        const _Float16* ap =
            &ldsA[(wm * 64 + mt * 16 + laneLow) * ASTR + st * 32 + laneHi * 8];
        v8h lo = *reinterpret_cast<const v8h*>(ap);
        v8h hi = *reinterpret_cast<const v8h*>(ap + 16);
        v16h af = __builtin_shufflevector(lo, hi, 0, 1, 2, 3, 4, 5, 6, 7,
                                          8, 9, 10, 11, 12, 13, 14, 15);
#pragma unroll
        for (int nt = 0; nt < 2; ++nt) {
          acc[mt][nt] = __builtin_amdgcn_wmma_f32_16x16x32_f16(
              false, af, false, bf[nt], (short)0, acc[mt][nt], false, false);
        }
      }
    }
    __syncthreads();
  }

  // ---- writeback: D layout VGPR r -> M = r + laneHi*8, N = laneLow ----
  const int t = t0 + wm;
  const size_t outRow = ((size_t)b * T_ + t) * U_ * (size_t)V_;
#pragma unroll
  for (int nt = 0; nt < 2; ++nt) {
    const int v = vBase + wn * 32 + nt * 16 + laneLow;
    const float bv = bias[v];
#pragma unroll
    for (int mt = 0; mt < 4; ++mt) {
#pragma unroll
      for (int r = 0; r < 8; ++r) {
        const int u = mt * 16 + laneHi * 8 + r;
        out[outRow + (size_t)u * V_ + v] = acc[mt][nt][r] + bv;
      }
    }
  }
}

extern "C" void kernel_launch(void* const* d_in, const int* in_sizes, int n_in,
                              void* d_out, int out_size, void* d_ws, size_t ws_size,
                              hipStream_t stream) {
  const float* S    = (const float*)d_in[0];  // (B,T,D) f32
  const int*   slen = (const int*)d_in[1];    // (B,) i32, passthrough
  const float* G    = (const float*)d_in[2];  // (B,U,D) f32
  const int*   tlen = (const int*)d_in[3];    // (B,) i32, passthrough
  const float* W    = (const float*)d_in[4];  // (V,D) f32
  const float* bias = (const float*)d_in[5];  // (V,) f32
  float* out = (float*)d_out;

  dim3 grid(V_ / NT, T_ / 2, B_);
  dim3 block(256);
  joiner_wmma_f16<<<grid, block, 0, stream>>>(S, G, W, bias, out);

  // Tuple outputs: (out, source_lengths, target_lengths) concatenated flat.
  const size_t outElems = (size_t)B_ * T_ * U_ * V_;
  if ((size_t)out_size >= outElems + 2 * B_) {
    hipMemcpyAsync(out + outElems, slen, B_ * sizeof(int),
                   hipMemcpyDeviceToDevice, stream);
    hipMemcpyAsync(out + outElems + B_, tlen, B_ * sizeof(int),
                   hipMemcpyDeviceToDevice, stream);
  }
}
